// ContextQueryAttentionLayer_32933809226479
// MI455X (gfx1250) — matrix operand order repaired and checked by reading, not compiled, not run on hardware
//
#include <hip/hip_runtime.h>
#include <hip/hip_bf16.h>

// ---------------- problem constants ----------------
constexpr int B_  = 32;
constexpr int LC_ = 1024;
constexpr int LQ_ = 256;
constexpr int D_  = 512;
#define NEGV (-1.0e10f)

// ---------------- CDNA5 WMMA types ----------------
typedef __attribute__((ext_vector_type(8)))  __bf16 v8bf;
typedef __attribute__((ext_vector_type(16))) __bf16 v16bf;
typedef __attribute__((ext_vector_type(8)))  float  v8f;

__device__ inline v16bf cat16(v8bf lo, v8bf hi) {
  return __builtin_shufflevector(lo, hi, 0,1,2,3,4,5,6,7,8,9,10,11,12,13,14,15);
}

__device__ inline __bf16 f2bf(float f) {
  union { float f; unsigned u; } in; in.f = f;
  unsigned u = in.u;
  u += 0x7fffu + ((u >> 16) & 1u);               // round-to-nearest-even
  union { unsigned short s; __bf16 b; } out; out.s = (unsigned short)(u >> 16);
  return out.b;
}

// A fragment (16xK row-major, 16x32 bf16 slice at kbase):
// lanes 0-15: M=lane, K in {kbase..+7, kbase+16..+23}
// lanes 16-31: M=lane-16, K in {kbase+8..+15, kbase+24..+31}
__device__ inline v16bf load_a_frag(const __bf16* A, int pitch, int kbase, int lane) {
  int m  = lane & 15;
  int kb = (lane < 16) ? 0 : 8;
  const __bf16* p = A + (size_t)m * pitch + kbase + kb;
  v8bf lo = *(const v8bf*)p;
  v8bf hi = *(const v8bf*)(p + 16);
  return cat16(lo, hi);
}

// B fragment (K x 16) read from BT (n-major: BT[n][k], row pitch `pitch`):
// lane holds column n = lane&15, K in {kbase..+7, kbase+16..+23} (+8 for upper half-wave)
__device__ inline v16bf load_b_frag(const __bf16* BT, int pitch, int kbase, int lane) {
  int n  = lane & 15;
  int kb = (lane < 16) ? 0 : 8;
  const __bf16* p = BT + (size_t)n * pitch + kbase + kb;
  v8bf lo = *(const v8bf*)p;
  v8bf hi = *(const v8bf*)(p + 16);
  return cat16(lo, hi);
}

__device__ inline v8f wmma_bf16(v16bf a, v16bf b, v8f c) {
  return __builtin_amdgcn_wmma_f32_16x16x32_bf16(false, a, false, b, (short)0, c, false, false);
}

// ---------------- K1a: cw = C@wc, qw = Q@wq (wave per row) ----------------
__global__ __launch_bounds__(256)
void k_rowdots(const float* __restrict__ C, const float* __restrict__ Q,
               const float* __restrict__ W0, float* __restrict__ cw, float* __restrict__ qw) {
  int wid  = (blockIdx.x * blockDim.x + threadIdx.x) >> 5;
  int lane = threadIdx.x & 31;
  const int nC = B_ * LC_;
  float s = 0.f;
  bool isC = wid < nC;
  if (!isC && wid - nC >= B_ * LQ_) return;
  const float* row = isC ? (C + (size_t)wid * D_) : (Q + (size_t)(wid - nC) * D_);
  const float* w   = isC ? W0 : (W0 + D_);
  for (int d = lane; d < D_; d += 32) s += row[d] * w[d];
  #pragma unroll
  for (int off = 16; off; off >>= 1) s += __shfl_xor(s, off, 32);
  if (lane == 0) { if (isC) cw[wid] = s; else qw[wid - nC] = s; }
}

// ---------------- K1b: Cmb = bf16(C*wm) row-major; Ct = bf16(C^T) ----------------
__global__ __launch_bounds__(256)
void k_prepC(const float* __restrict__ C, const float* __restrict__ W0,
             __bf16* __restrict__ Cmb, __bf16* __restrict__ Ct) {
  __shared__ float tile[32][33];
  int b = blockIdx.z, d0 = blockIdx.x * 32, i0 = blockIdx.y * 32;
  int x = threadIdx.x, y0 = threadIdx.y;            // 32 x 8
  const float* Cb = C + (size_t)b * LC_ * D_;
  #pragma unroll
  for (int ry = 0; ry < 4; ry++) {
    int y = y0 + ry * 8;
    float v = Cb[(size_t)(i0 + y) * D_ + d0 + x];
    tile[y][x] = v;
    Cmb[((size_t)b * LC_ + i0 + y) * D_ + d0 + x] = f2bf(v * W0[2 * D_ + d0 + x]);
  }
  __syncthreads();
  #pragma unroll
  for (int ry = 0; ry < 4; ry++) {
    int y = y0 + ry * 8;
    Ct[((size_t)b * D_ + d0 + y) * LC_ + i0 + x] = f2bf(tile[x][y]);
  }
}

// ---------------- K1c: Qb = bf16(Q) row-major; Qt = bf16(Q^T) ----------------
__global__ __launch_bounds__(256)
void k_prepQ(const float* __restrict__ Q, __bf16* __restrict__ Qb, __bf16* __restrict__ Qt) {
  __shared__ float tile[32][33];
  int b = blockIdx.z, d0 = blockIdx.x * 32, j0 = blockIdx.y * 32;
  int x = threadIdx.x, y0 = threadIdx.y;
  const float* Qbase = Q + (size_t)b * LQ_ * D_;
  #pragma unroll
  for (int ry = 0; ry < 4; ry++) {
    int y = y0 + ry * 8;
    float v = Qbase[(size_t)(j0 + y) * D_ + d0 + x];
    tile[y][x] = v;
    Qb[((size_t)b * LQ_ + j0 + y) * D_ + d0 + x] = f2bf(v);
  }
  __syncthreads();
  #pragma unroll
  for (int ry = 0; ry < 4; ry++) {
    int y = y0 + ry * 8;
    Qt[((size_t)b * D_ + d0 + y) * LQ_ + j0 + x] = f2bf(tile[x][y]);
  }
}

// ---------------- K2: scores + row softmax (32-row M-tile) ----------------
// Per (b, 32-row i-tile): S = Cm @ Q^T (+cw+qw); write raw S (f32) and Prow (bf16).
// A-panel (32 KB) and S buffer (~32.9 KB) share LDS via a union: A-panel is dead
// after the K-loop; phases separated by __syncthreads().
__global__ __launch_bounds__(256)
void k_score(const __bf16* __restrict__ Cmb, const __bf16* __restrict__ Qb,
             const float* __restrict__ cw, const float* __restrict__ qw,
             const int* __restrict__ qmask,
             float* __restrict__ Sws, __bf16* __restrict__ Prow) {
  __shared__ union {
    __bf16 apanel[32 * 512];   // 32 KB
    float  sbuf[32][257];      // padded: conflict-free column access
  } sh;
  int b = blockIdx.y, i0 = blockIdx.x * 32;
  int t = threadIdx.x, lane = t & 31, w = t >> 5;

  { // stage A-panel (Cmb rows i0..i0+31, full K=512) = 2048 uint4
    const uint4* src = (const uint4*)(Cmb + ((size_t)b * LC_ + i0) * D_);
    uint4* dst = (uint4*)sh.apanel;
    #pragma unroll
    for (int k = t; k < 2048; k += 256) dst[k] = src[k];
  }
  __syncthreads();

  v8f acc[2][2] = {};                               // [M-tile][N-tile]
  int n0 = w * 32, n1 = n0 + 16;
  const __bf16* QbB = Qb + (size_t)b * LQ_ * D_;    // BT for S: Q row-major (n=j, k=d)
  #pragma unroll 4
  for (int ks = 0; ks < 16; ks++) {
    int kb = ks * 32;
    v16bf a0 = load_a_frag(sh.apanel,            512, kb, lane);
    v16bf a1 = load_a_frag(sh.apanel + 16 * 512, 512, kb, lane);
    v16bf b0 = load_b_frag(QbB + (size_t)n0 * D_, D_, kb, lane);
    acc[0][0] = wmma_bf16(a0, b0, acc[0][0]);
    acc[1][0] = wmma_bf16(a1, b0, acc[1][0]);
    v16bf b1 = load_b_frag(QbB + (size_t)n1 * D_, D_, kb, lane);
    acc[0][1] = wmma_bf16(a0, b1, acc[0][1]);
    acc[1][1] = wmma_bf16(a1, b1, acc[1][1]);
  }
  __syncthreads();   // A-panel dead; switch union to S-buffer phase

  int mo  = (lane < 16) ? 0 : 8;
  int nn0 = n0 + (lane & 15), nn1 = n1 + (lane & 15);
  float qw0 = qw[b * LQ_ + nn0], qw1 = qw[b * LQ_ + nn1];
  #pragma unroll
  for (int mt = 0; mt < 2; mt++) {
    #pragma unroll
    for (int r = 0; r < 8; r++) {
      int m = mt * 16 + r + mo, i = i0 + m;
      float base = cw[b * LC_ + i];
      float s0 = acc[mt][0][r] + base + qw0;
      float s1 = acc[mt][1][r] + base + qw1;
      Sws[((size_t)b * LC_ + i) * LQ_ + nn0] = s0;
      Sws[((size_t)b * LC_ + i) * LQ_ + nn1] = s1;
      sh.sbuf[m][nn0] = s0; sh.sbuf[m][nn1] = s1;
    }
  }
  __syncthreads();

  // row softmax: 16 threads (half-wave) per row; 2 rows per thread-group
  const int* qm = qmask + b * LQ_;
  int sj = t & 15;
  #pragma unroll
  for (int rr = 0; rr < 2; rr++) {
    int r = (t >> 4) + rr * 16;
    float vbuf[16];
    float mx = -3.0e38f;
    #pragma unroll
    for (int idx = 0; idx < 16; idx++) {
      int c = sj + idx * 16;
      float v = (qm[c] == 1) ? NEGV : sh.sbuf[r][c];
      vbuf[idx] = v;
      mx = fmaxf(mx, v);
    }
    #pragma unroll
    for (int off = 8; off; off >>= 1) mx = fmaxf(mx, __shfl_xor(mx, off, 32));
    float sum = 0.f;
    #pragma unroll
    for (int idx = 0; idx < 16; idx++) { float e = __expf(vbuf[idx] - mx); vbuf[idx] = e; sum += e; }
    #pragma unroll
    for (int off = 8; off; off >>= 1) sum += __shfl_xor(sum, off, 32);
    float inv = 1.0f / sum;
    #pragma unroll
    for (int idx = 0; idx < 16; idx++) {
      int c = sj + idx * 16;
      Prow[((size_t)b * LC_ + i0 + r) * LQ_ + c] = f2bf(vbuf[idx] * inv);
    }
  }
}

// ---------------- K3: column softmax (unnormalized) -> PcolT, 1/colsum ----------------
__global__ __launch_bounds__(256)
void k_col(const float* __restrict__ Sws, const int* __restrict__ cmask,
           __bf16* __restrict__ PcolT, float* __restrict__ colinv) {
  int gw = (blockIdx.x * blockDim.x + threadIdx.x) >> 5;   // b*LQ + j
  int lane = threadIdx.x & 31;
  int b = gw / LQ_, j = gw - b * LQ_;
  const float* Scol = Sws + (size_t)b * LC_ * LQ_ + j;
  const int* cm = cmask + b * LC_;
  float mx = -3.0e38f;
  for (int i = lane; i < LC_; i += 32) {
    float v = (cm[i] == 1) ? NEGV : Scol[(size_t)i * LQ_];
    mx = fmaxf(mx, v);
  }
  #pragma unroll
  for (int off = 16; off; off >>= 1) mx = fmaxf(mx, __shfl_xor(mx, off, 32));
  float sum = 0.f;
  __bf16* Pc = PcolT + ((size_t)b * LQ_ + j) * LC_;
  for (int i = lane; i < LC_; i += 32) {
    float v = (cm[i] == 1) ? NEGV : Scol[(size_t)i * LQ_];
    float e = __expf(v - mx);
    sum += e;
    Pc[i] = f2bf(e);
  }
  #pragma unroll
  for (int off = 16; off; off >>= 1) sum += __shfl_xor(sum, off, 32);
  if (lane == 0) colinv[gw] = 1.0f / sum;
}

// ---------------- K4: Gt[b][d][j] = (1/colsum[j]) * sum_i PcolT[b][j][i]*C[b][i][d] ----
// A-panel staged with CDNA5 async global->LDS DMA (ASYNCcnt-tracked), then WMMA.
__global__ __launch_bounds__(256)
void k_g(const __bf16* __restrict__ PcolT, const __bf16* __restrict__ Ct,
         const float* __restrict__ colinv, __bf16* __restrict__ Gt) {
  __shared__ __bf16 Apanel[16 * 1024];                // 32 KB
  int b = blockIdx.z, j0 = blockIdx.y * 16, dblk = blockIdx.x;
  int t = threadIdx.x, lane = t & 31, w = t >> 5;
  { // async-stage PcolT rows j0..j0+15 (K = LC = 1024): 2048 x b128, 8 per thread
    const __bf16* gsrc = PcolT + ((size_t)b * LQ_ + j0) * LC_;
    #pragma unroll
    for (int it = 0; it < 8; it++) {
      unsigned idx = (unsigned)t + (unsigned)it * 256u;
      unsigned long long ga = (unsigned long long)(uintptr_t)gsrc + (unsigned long long)idx * 16u;
      unsigned la = (unsigned)(uintptr_t)(&Apanel[0]) + idx * 16u;   // LDS offset = addr[31:0]
      asm volatile("global_load_async_to_lds_b128 %0, %1, off"
                   :: "v"(la), "v"(ga) : "memory");
    }
    asm volatile("s_wait_asynccnt 0x0" ::: "memory");
  }
  __syncthreads();
  int nbase = (dblk * 8 + w) * 16;
  const __bf16* CtB = Ct + ((size_t)b * D_ + nbase) * LC_;
  v8f acc = {};
  #pragma unroll 4
  for (int ks = 0; ks < 32; ks++) {
    v16bf a  = load_a_frag(Apanel, 1024, ks * 32, lane);
    v16bf bb = load_b_frag(CtB, LC_, ks * 32, lane);
    acc = wmma_bf16(a, bb, acc);
  }
  int mo = (lane < 16) ? 0 : 8;
  int n  = nbase + (lane & 15);
  #pragma unroll
  for (int r = 0; r < 8; r++) {
    int j = j0 + r + mo;
    Gt[((size_t)b * D_ + n) * LQ_ + j] = f2bf(acc[r] * colinv[b * LQ_ + j]);
  }
}

// ---------------- K5: [A|Bt] = Prow @ [Q|G]; fused output concat (32-row M-tile) ----
// Grid y splits the 512-wide N dimension in halves; each wave owns 2 M-tiles x 2 N-tiles
// per side (8 accumulators), halving B-panel L2 traffic vs 16-row tiles.
__global__ __launch_bounds__(256)
void k_final(const __bf16* __restrict__ Prow, const __bf16* __restrict__ Qt,
             const __bf16* __restrict__ Gt, const float* __restrict__ C,
             float* __restrict__ Out) {
  __shared__ __bf16 Apanel[32 * 256];                 // 16 KB
  int b = blockIdx.z, i0 = blockIdx.x * 32;
  int dhalf = blockIdx.y * 256;
  int t = threadIdx.x, lane = t & 31, w = t >> 5;
  { // stage Prow rows i0..i0+31 (K = LQ = 256) = 1024 uint4
    const uint4* src = (const uint4*)(Prow + ((size_t)b * LC_ + i0) * LQ_);
    uint4* dst = (uint4*)Apanel;
    #pragma unroll
    for (int k = t; k < 1024; k += 256) dst[k] = src[k];
  }
  __syncthreads();
  v8f accA[2][2] = {}; v8f accB[2][2] = {};           // [M-tile][N-tile]
  int n0 = dhalf + w * 32, n1 = n0 + 16;
  const __bf16* QtB = Qt + (size_t)b * D_ * LQ_;
  const __bf16* GtB = Gt + (size_t)b * D_ * LQ_;
  #pragma unroll
  for (int ks = 0; ks < 8; ks++) {
    int kb = ks * 32;
    v16bf a0 = load_a_frag(Apanel,            256, kb, lane);
    v16bf a1 = load_a_frag(Apanel + 16 * 256, 256, kb, lane);
    v16bf bq0 = load_b_frag(QtB + (size_t)n0 * LQ_, LQ_, kb, lane);
    accA[0][0] = wmma_bf16(a0, bq0, accA[0][0]);
    accA[1][0] = wmma_bf16(a1, bq0, accA[1][0]);
    v16bf bq1 = load_b_frag(QtB + (size_t)n1 * LQ_, LQ_, kb, lane);
    accA[0][1] = wmma_bf16(a0, bq1, accA[0][1]);
    accA[1][1] = wmma_bf16(a1, bq1, accA[1][1]);
    v16bf bg0 = load_b_frag(GtB + (size_t)n0 * LQ_, LQ_, kb, lane);
    accB[0][0] = wmma_bf16(a0, bg0, accB[0][0]);
    accB[1][0] = wmma_bf16(a1, bg0, accB[1][0]);
    v16bf bg1 = load_b_frag(GtB + (size_t)n1 * LQ_, LQ_, kb, lane);
    accB[0][1] = wmma_bf16(a0, bg1, accB[0][1]);
    accB[1][1] = wmma_bf16(a1, bg1, accB[1][1]);
  }
  int mo = (lane < 16) ? 0 : 8;
  const float* Cb = C + (size_t)b * LC_ * D_;
  float* Ob = Out + (size_t)b * LC_ * (4 * D_);
  #pragma unroll
  for (int mt = 0; mt < 2; mt++) {
    #pragma unroll
    for (int nt = 0; nt < 2; nt++) {
      int d = (nt ? n1 : n0) + (lane & 15);
      #pragma unroll
      for (int r = 0; r < 8; r++) {
        int i = i0 + mt * 16 + r + mo;
        float c  = Cb[(size_t)i * D_ + d];
        float av = accA[mt][nt][r];
        float bv = accB[mt][nt][r];
        float* orow = Ob + (size_t)i * (4 * D_);
        orow[d]          = c;
        orow[D_ + d]     = av;
        orow[2 * D_ + d] = c * av;
        orow[3 * D_ + d] = c * bv;
      }
    }
  }
}

// ---------------- host launcher ----------------
extern "C" void kernel_launch(void* const* d_in, const int* in_sizes, int n_in,
                              void* d_out, int out_size, void* d_ws, size_t ws_size,
                              hipStream_t stream) {
  (void)in_sizes; (void)n_in; (void)out_size; (void)ws_size;
  const float* C  = (const float*)d_in[0];
  const float* Q  = (const float*)d_in[1];
  const float* W0 = (const float*)d_in[2];
  const int* cmask = (const int*)d_in[3];
  const int* qmask = (const int*)d_in[4];
  float* Out = (float*)d_out;

  char* wp = (char*)d_ws;
  size_t off = 0;
  auto take = [&](size_t bytes) -> char* {
    char* p = wp + off;
    off += (bytes + 255) & ~(size_t)255;
    return p;
  };
  __bf16* Cmb   = (__bf16*)take((size_t)B_ * LC_ * D_ * 2);   // 32 MB
  __bf16* Ct    = (__bf16*)take((size_t)B_ * D_ * LC_ * 2);   // 32 MB
  __bf16* Qb    = (__bf16*)take((size_t)B_ * LQ_ * D_ * 2);   //  8 MB
  __bf16* Qt    = (__bf16*)take((size_t)B_ * D_ * LQ_ * 2);   //  8 MB
  float*  Sws   = (float*) take((size_t)B_ * LC_ * LQ_ * 4);  // 32 MB
  __bf16* Prow  = (__bf16*)take((size_t)B_ * LC_ * LQ_ * 2);  // 16 MB
  __bf16* PcolT = (__bf16*)take((size_t)B_ * LQ_ * LC_ * 2);  // 16 MB
  __bf16* Gt    = (__bf16*)take((size_t)B_ * D_ * LQ_ * 2);   //  8 MB
  float*  cw    = (float*) take((size_t)B_ * LC_ * 4);
  float*  qw    = (float*) take((size_t)B_ * LQ_ * 4);
  float*  colinv= (float*) take((size_t)B_ * LQ_ * 4);

  // 1) row dots (wave per row): (B*LC + B*LQ) waves, 8 waves/block
  k_rowdots<<<(B_ * LC_ + B_ * LQ_) / 8, 256, 0, stream>>>(C, Q, W0, cw, qw);
  // 2) prep C: Cmb + Ct
  k_prepC<<<dim3(D_ / 32, LC_ / 32, B_), dim3(32, 8), 0, stream>>>(C, W0, Cmb, Ct);
  // 3) prep Q: Qb + Qt
  k_prepQ<<<dim3(D_ / 32, LQ_ / 32, B_), dim3(32, 8), 0, stream>>>(Q, Qb, Qt);
  // 4) scores + row softmax (32-row tiles)
  k_score<<<dim3(LC_ / 32, B_), 256, 0, stream>>>(Cmb, Qb, cw, qw, qmask, Sws, Prow);
  // 5) column softmax
  k_col<<<(B_ * LQ_) / 8, 256, 0, stream>>>(Sws, cmask, PcolT, colinv);
  // 6) G = S_col^T @ C (transposed output), async-LDS staged
  k_g<<<dim3(4, LQ_ / 16, B_), 256, 0, stream>>>(PcolT, Ct, colinv, Gt);
  // 7) final fused matmul + concat (32-row tiles, N split in halves)
  k_final<<<dim3(LC_ / 32, 2, B_), 256, 0, stream>>>(Prow, Qt, Gt, C, Out);
}
